// DCRNN_20280835571773
// MI455X (gfx1250) — compile-verified, matrix operand order checked by reading, not run on hardware
//
#include <hip/hip_runtime.h>
#include <hip/hip_bf16.h>

// DCRNN on MI455X (gfx1250). All GEMMs -> V_WMMA_F32_16X16X32_BF16, f32 accum.
// Weights/adjacency pre-converted to zero-padded bf16; activations carried in
// bf16 (z transposed for the diffusion B operand); h kept in f32 for the GRU
// blend with a bf16 shadow for matrix use. Diffusion stages the shared adj
// tile through LDS (ds_load_b128) to kill the 8x redundant global fetch.
// B=64, T=6, N=714, D=16, HID=128.

#define BATCH   64
#define TSTEPS  6
#define NNODES  714
#define DIN     16
#define HIDDEN  128
#define BN      (BATCH * NNODES)   // 45696 = 16 * 2856

#define ADJ_ROWS 720               // 714 padded up to 16-row tiles
#define LDA      736               // adj leading dim, covers K<=735, mult of 8
#define LDZT     736               // zT leading dim over nodes, zero padded

typedef __attribute__((ext_vector_type(16))) __bf16 v16bf;
typedef __attribute__((ext_vector_type(8)))  __bf16 v8bf;
typedef __attribute__((ext_vector_type(8)))  float  v8f;

static __device__ __forceinline__ v8f wmma_bf16(v16bf a, v16bf b, v8f c) {
    return __builtin_amdgcn_wmma_f32_16x16x32_bf16(
        /*neg_a=*/false, a, /*neg_b=*/false, b,
        /*c_mod=*/(short)0, c, /*reuse_a=*/false, /*reuse_b=*/false);
}

// ---------------------------------------------------------------------------
// bf16 fragment load (CDNA5 16-bit 16x32 layout, ISA 7.12.2):
//   lane<16 : i = lane,    K in {0..7, 16..23}
//   lane>=16: i = lane-16, K in {8..15, 24..31}
// element (i,k) = src[i*ld + k]. Works for A tiles (i = M row) and for
// B tiles whenever the column of B is contiguous in memory (W^T of a
// row-major W, or a pre-transposed activation). Source must be 16B aligned
// (all our lds/ld offsets are multiples of 8 elements) and zero-padded, so
// loads are unguarded vector loads: 2x b128 per fragment, no cvt, no select.
// ---------------------------------------------------------------------------
static __device__ __forceinline__ v16bf frag_bf16(const __bf16* src, int ld) {
    const int l  = threadIdx.x & 31;
    const int i  = l & 15;
    const int kh = (l >> 4) * 8;
    const __bf16* p = src + (size_t)i * ld + kh;
    v8bf lo = *(const v8bf*)(p);
    v8bf hi = *(const v8bf*)(p + 16);
    return __builtin_shufflevector(lo, hi, 0, 1, 2, 3, 4, 5, 6, 7,
                                           8, 9, 10, 11, 12, 13, 14, 15);
}

// ---------------------------------------------------------------------------
// one-time converts / zero fills
// ---------------------------------------------------------------------------
__global__ void f32_to_bf16(const float* __restrict__ s, __bf16* __restrict__ d, int n) {
    int i = blockIdx.x * blockDim.x + threadIdx.x;
    if (i < n) d[i] = (__bf16)s[i];
}

__global__ void zero_f32(float* __restrict__ p, int n) {
    int i = blockIdx.x * blockDim.x + threadIdx.x;
    if (i < n) p[i] = 0.0f;
}

__global__ void zero_bf16(__bf16* __restrict__ p, int n) {
    int i = blockIdx.x * blockDim.x + threadIdx.x;
    if (i < n) p[i] = (__bf16)0.0f;
}

__global__ void conv_pad_adj(const float* __restrict__ adj, __bf16* __restrict__ d) {
    int idx = blockIdx.x * blockDim.x + threadIdx.x;
    if (idx >= ADJ_ROWS * LDA) return;
    int r = idx / LDA, c = idx - r * LDA;
    float v = (r < NNODES && c < NNODES) ? adj[(size_t)r * NNODES + c] : 0.0f;
    d[idx] = (__bf16)v;
}

__global__ void conv_pad_outw(const float* __restrict__ w, __bf16* __restrict__ d) {
    int idx = blockIdx.x * blockDim.x + threadIdx.x;
    if (idx >= ADJ_ROWS * HIDDEN) return;
    int r = idx / HIDDEN, c = idx - r * HIDDEN;
    float v = (r < NNODES) ? w[(size_t)r * HIDDEN + c] : 0.0f;
    d[idx] = (__bf16)v;
}

// ---------------------------------------------------------------------------
// z = relu([x_t | h] @ fc_w^T + fc_b), stored TRANSPOSED in bf16:
//   zT[b][hcol][node]  (leading dim LDZT, node pad pre-zeroed)
// grid.x = BN/16; block 256 (8 waves; wave w owns hid cols [16w,16w+16))
// ---------------------------------------------------------------------------
__global__ void fc_relu_kernel(const float*  __restrict__ x,     // [B,T,N,D] f32
                               const __bf16* __restrict__ hbf,   // [BN,HIDDEN]
                               const __bf16* __restrict__ fcw,   // [HIDDEN,D+HIDDEN]
                               const float*  __restrict__ fc_b,  // [HIDDEN]
                               __bf16*       __restrict__ zT,    // [B,HIDDEN,LDZT]
                               int t) {
    const int rowBase = blockIdx.x * 16;
    const int colBase = (threadIdx.x >> 5) * 16;
    const int l  = threadIdx.x & 31;
    const int i  = l & 15;
    const int kh = (l >> 4) * 8;

    v8f acc = {};

    // ---- x part: K = 0..15 (upper K-half zero) ----
    {
        int r = rowBase + i;
        int b = r / NNODES;
        int n = r - b * NNODES;
        const float* xp = x + (((size_t)b * TSTEPS + t) * NNODES + n) * DIN + kh;
        v16bf a;
#pragma unroll
        for (int e = 0; e < 8; ++e) a[e] = (__bf16)xp[e];
#pragma unroll
        for (int e = 0; e < 8; ++e) a[8 + e] = (__bf16)0.0f;
        v8bf wlo = *(const v8bf*)(fcw + (size_t)(colBase + i) * (DIN + HIDDEN) + kh);
        v8bf zer = {};
        v16bf bf = __builtin_shufflevector(wlo, zer, 0, 1, 2, 3, 4, 5, 6, 7,
                                                     8, 9, 10, 11, 12, 13, 14, 15);
        acc = wmma_bf16(a, bf, acc);
    }

    // ---- h part: K = 16..143 -> h columns 0..127 ----
#pragma unroll
    for (int k0 = 0; k0 < HIDDEN; k0 += 32) {
        v16bf a  = frag_bf16(hbf + (size_t)rowBase * HIDDEN + k0, HIDDEN);
        v16bf bf = frag_bf16(fcw + (size_t)colBase * (DIN + HIDDEN) + DIN + k0,
                             DIN + HIDDEN);
        acc = wmma_bf16(a, bf, acc);
    }

    const float bias = fc_b[colBase + i];
#pragma unroll
    for (int v = 0; v < 8; ++v) {
        int row = rowBase + ((l < 16) ? v : v + 8);
        int b = row / NNODES;
        int n = row - b * NNODES;
        float val = acc[v] + bias;
        val = val > 0.0f ? val : 0.0f;
        zT[((size_t)b * HIDDEN + colBase + i) * LDZT + n] = (__bf16)val;
    }
}

// ---------------------------------------------------------------------------
// z2[b] = adj @ z[b] : A = adj tile staged through LDS (shared by all 8
// waves), B = zT rows (contiguous). Everything bf16, zero-padded -> no guards
// on loads. grid = (45, BATCH); block 256.
// ---------------------------------------------------------------------------
__global__ void diffusion_kernel(const __bf16* __restrict__ adjbf, // [720,LDA]
                                 const __bf16* __restrict__ zT,    // [B,HIDDEN,LDZT]
                                 __bf16*       __restrict__ z2bf) { // [BN,HIDDEN]
    __shared__ __align__(16) __bf16 atile[16 * 32];

    const int rowBase = blockIdx.x * 16;
    const int b       = blockIdx.y;
    const int colBase = (threadIdx.x >> 5) * 16;
    const int l = threadIdx.x & 31;
    const int i = l & 15;

    const __bf16* zTb = zT + (size_t)b * HIDDEN * LDZT + (size_t)colBase * LDZT;

    v8f acc = {};
    for (int k0 = 0; k0 < NNODES; k0 += 32) {
        __syncthreads();
        {   // cooperative stage: 512 bf16 = 1KB, 2 elems per thread
            int r = threadIdx.x >> 4;          // 0..15
            int c = (threadIdx.x & 15) * 2;    // 0,2,..,30
            const __bf16* s = adjbf + (size_t)(rowBase + r) * LDA + k0 + c;
            *(uint32_t*)&atile[r * 32 + c] = *(const uint32_t*)s;
        }
        __syncthreads();
        __builtin_prefetch(zTb + k0 + 32, 0, 0);  // next zT chunk
        v16bf a  = frag_bf16(atile, 32);          // ds_load_b128 x2
        v16bf bf = frag_bf16(zTb + k0, LDZT);
        acc = wmma_bf16(a, bf, acc);
    }

    __bf16* out = z2bf + (size_t)b * NNODES * HIDDEN;
#pragma unroll
    for (int v = 0; v < 8; ++v) {
        int row = rowBase + ((l < 16) ? v : v + 8);
        if (row < NNODES)
            out[(size_t)row * HIDDEN + colBase + i] = (__bf16)acc[v];
    }
}

// ---------------------------------------------------------------------------
// Fused GRU cell. A operands (z2, h) and B operands (w_ih, w_hh) all bf16.
// h blend read from the f32 master copy; writes f32 master + bf16 shadow.
// grid.x = BN/16; block 256.
// ---------------------------------------------------------------------------
__global__ void gru_kernel(const __bf16* __restrict__ z2bf,  // [BN,HIDDEN]
                           const __bf16* __restrict__ hbf,   // [BN,HIDDEN]
                           const float*  __restrict__ hf32,  // [BN,HIDDEN]
                           const __bf16* __restrict__ wih,   // [3H,H]
                           const __bf16* __restrict__ whh,   // [3H,H]
                           const float*  __restrict__ b_ih,  // [3H]
                           const float*  __restrict__ b_hh,  // [3H]
                           float*        __restrict__ hof,   // [BN,HIDDEN]
                           __bf16*       __restrict__ hob) { // [BN,HIDDEN]
    const int rowBase = blockIdx.x * 16;
    const int colBase = (threadIdx.x >> 5) * 16;
    const int l = threadIdx.x & 31;
    const int i = l & 15;

    v8f ir = {}, iz = {}, in_ = {};
    v8f hr = {}, hz = {}, hn  = {};

#pragma unroll
    for (int k0 = 0; k0 < HIDDEN; k0 += 32) {
        v16bf az = frag_bf16(z2bf + (size_t)rowBase * HIDDEN + k0, HIDDEN);
        v16bf ah = frag_bf16(hbf  + (size_t)rowBase * HIDDEN + k0, HIDDEN);
        v16bf wr = frag_bf16(wih + (size_t)(0 * HIDDEN + colBase) * HIDDEN + k0, HIDDEN);
        v16bf wz = frag_bf16(wih + (size_t)(1 * HIDDEN + colBase) * HIDDEN + k0, HIDDEN);
        v16bf wn = frag_bf16(wih + (size_t)(2 * HIDDEN + colBase) * HIDDEN + k0, HIDDEN);
        v16bf vr = frag_bf16(whh + (size_t)(0 * HIDDEN + colBase) * HIDDEN + k0, HIDDEN);
        v16bf vz = frag_bf16(whh + (size_t)(1 * HIDDEN + colBase) * HIDDEN + k0, HIDDEN);
        v16bf vn = frag_bf16(whh + (size_t)(2 * HIDDEN + colBase) * HIDDEN + k0, HIDDEN);
        ir  = wmma_bf16(az, wr, ir);
        iz  = wmma_bf16(az, wz, iz);
        in_ = wmma_bf16(az, wn, in_);
        hr  = wmma_bf16(ah, vr, hr);
        hz  = wmma_bf16(ah, vz, hz);
        hn  = wmma_bf16(ah, vn, hn);
    }

    const int c = colBase + i;
    const float bir = b_ih[0 * HIDDEN + c], bhr = b_hh[0 * HIDDEN + c];
    const float biz = b_ih[1 * HIDDEN + c], bhz = b_hh[1 * HIDDEN + c];
    const float bin = b_ih[2 * HIDDEN + c], bhn = b_hh[2 * HIDDEN + c];

#pragma unroll
    for (int v = 0; v < 8; ++v) {
        int row = rowBase + ((l < 16) ? v : v + 8);
        float r = 1.0f / (1.0f + __expf(-(ir[v] + bir + hr[v] + bhr)));
        float u = 1.0f / (1.0f + __expf(-(iz[v] + biz + hz[v] + bhz)));
        float cc = tanhf(in_[v] + bin + r * (hn[v] + bhn));
        float ho = hf32[(size_t)row * HIDDEN + c];
        float hv = (1.0f - u) * cc + u * ho;
        hof[(size_t)row * HIDDEN + c] = hv;
        hob[(size_t)row * HIDDEN + c] = (__bf16)hv;
    }
}

// ---------------------------------------------------------------------------
// y = h @ out_w^T + out_b : [BN, NNODES] f32 output.
// grid = (BN/128 = 357, 45); block 256 (wave w owns rows rowBase + 16w).
// ---------------------------------------------------------------------------
__global__ void out_kernel(const __bf16* __restrict__ hbf,   // [BN,HIDDEN]
                           const __bf16* __restrict__ outw,  // [720,HIDDEN] padded
                           const float*  __restrict__ out_b, // [N]
                           float*        __restrict__ y) {
    const int rowBase = blockIdx.x * 128 + (threadIdx.x >> 5) * 16;
    const int colBase = blockIdx.y * 16;
    const int l = threadIdx.x & 31;
    const int i = l & 15;

    v8f acc = {};
#pragma unroll
    for (int k0 = 0; k0 < HIDDEN; k0 += 32) {
        v16bf a  = frag_bf16(hbf  + (size_t)rowBase * HIDDEN + k0, HIDDEN);
        v16bf bf = frag_bf16(outw + (size_t)colBase * HIDDEN + k0, HIDDEN);
        acc = wmma_bf16(a, bf, acc);
    }

    const int col = colBase + i;
    if (col < NNODES) {
        const float bias = out_b[col];
#pragma unroll
        for (int v = 0; v < 8; ++v) {
            int row = rowBase + ((l < 16) ? v : v + 8);
            y[(size_t)row * NNODES + col] = acc[v] + bias;
        }
    }
}

// ---------------------------------------------------------------------------
extern "C" void kernel_launch(void* const* d_in, const int* in_sizes, int n_in,
                              void* d_out, int out_size, void* d_ws, size_t ws_size,
                              hipStream_t stream) {
    (void)in_sizes; (void)n_in; (void)out_size; (void)ws_size;
    const float* x     = (const float*)d_in[0];
    const float* adj   = (const float*)d_in[1];
    const float* fc_w  = (const float*)d_in[2];
    const float* fc_b  = (const float*)d_in[3];
    const float* w_ih  = (const float*)d_in[4];
    const float* w_hh  = (const float*)d_in[5];
    const float* b_ih  = (const float*)d_in[6];
    const float* b_hh  = (const float*)d_in[7];
    const float* out_w = (const float*)d_in[8];
    const float* out_b = (const float*)d_in[9];
    float* y = (float*)d_out;

    auto align256 = [](size_t v) { return (v + 255) & ~(size_t)255; };
    char* ws = (char*)d_ws;
    size_t off = 0;
    auto carve = [&](size_t bytes) { char* p = ws + off; off += align256(bytes); return p; };

    const size_t hElems  = (size_t)BN * HIDDEN;
    float*  hf32A = (float*)carve(hElems * 4);
    float*  hf32B = (float*)carve(hElems * 4);
    __bf16* hbfA  = (__bf16*)carve(hElems * 2);
    __bf16* hbfB  = (__bf16*)carve(hElems * 2);
    const size_t zTElems = (size_t)BATCH * HIDDEN * LDZT;
    __bf16* zT    = (__bf16*)carve(zTElems * 2);
    __bf16* z2bf  = (__bf16*)carve(hElems * 2);
    __bf16* adjbf = (__bf16*)carve((size_t)ADJ_ROWS * LDA * 2);
    __bf16* fcwbf = (__bf16*)carve((size_t)HIDDEN * (DIN + HIDDEN) * 2);
    __bf16* wihbf = (__bf16*)carve((size_t)3 * HIDDEN * HIDDEN * 2);
    __bf16* whhbf = (__bf16*)carve((size_t)3 * HIDDEN * HIDDEN * 2);
    __bf16* outwbf= (__bf16*)carve((size_t)ADJ_ROWS * HIDDEN * 2);

    const int TPB = 256;
    auto blocks = [&](size_t n) { return (int)((n + TPB - 1) / TPB); };

    // --- one-time (per call) converts & zero fills ---
    conv_pad_adj <<<blocks((size_t)ADJ_ROWS * LDA), TPB, 0, stream>>>(adj, adjbf);
    f32_to_bf16  <<<blocks((size_t)HIDDEN * (DIN + HIDDEN)), TPB, 0, stream>>>(fc_w, fcwbf, HIDDEN * (DIN + HIDDEN));
    f32_to_bf16  <<<blocks((size_t)3 * HIDDEN * HIDDEN), TPB, 0, stream>>>(w_ih, wihbf, 3 * HIDDEN * HIDDEN);
    f32_to_bf16  <<<blocks((size_t)3 * HIDDEN * HIDDEN), TPB, 0, stream>>>(w_hh, whhbf, 3 * HIDDEN * HIDDEN);
    conv_pad_outw<<<blocks((size_t)ADJ_ROWS * HIDDEN), TPB, 0, stream>>>(out_w, outwbf);
    zero_bf16    <<<blocks(zTElems), TPB, 0, stream>>>(zT, (int)zTElems);       // node pad stays 0
    zero_f32     <<<blocks(hElems), TPB, 0, stream>>>(hf32A, (int)hElems);      // h0 = 0
    zero_bf16    <<<blocks(hElems), TPB, 0, stream>>>(hbfA, (int)hElems);

    float*  hfin = hf32A;  float*  hfout = hf32B;
    __bf16* hbin = hbfA;   __bf16* hbout = hbfB;

    for (int t = 0; t < TSTEPS; ++t) {
        fc_relu_kernel<<<BN / 16, TPB, 0, stream>>>(x, hbin, fcwbf, fc_b, zT, t);

        dim3 gdiff((NNODES + 15) / 16, BATCH);
        diffusion_kernel<<<gdiff, TPB, 0, stream>>>(adjbf, zT, z2bf);

        gru_kernel<<<BN / 16, TPB, 0, stream>>>(z2bf, hbin, hfin, wihbf, whhbf,
                                                b_ih, b_hh, hfout, hbout);

        float*  tf = hfin; hfin = hfout; hfout = tf;
        __bf16* tb = hbin; hbin = hbout; hbout = tb;
    }

    dim3 gout(BN / 128, (NNODES + 15) / 16);
    out_kernel<<<gout, TPB, 0, stream>>>(hbin, outwbf, out_b, y);
}
